// S4Layer_18803366821989
// MI455X (gfx1250) — matrix operand order repaired and checked by reading, not compiled, not run on hardware
//
#include <hip/hip_runtime.h>
#include <hip/hip_bf16.h>
#include <math.h>

typedef __attribute__((ext_vector_type(16))) __bf16 v16bf;
typedef __attribute__((ext_vector_type(8)))  __bf16 v8bf;
typedef __attribute__((ext_vector_type(8)))  float  v8f;

constexpr int Bdim = 8, Hdim = 256, Ldim = 2048, Ndim = 128;
constexpr int NDELTA = 254;               // distinct (l0 - t0)/16 values
constexpr int DELTA_MIN = -2016;          // l0 in [0,2032], t0 in [0,2016]

static __device__ __forceinline__ unsigned short f2bf_bits(float f) {
  unsigned int u = __builtin_bit_cast(unsigned int, f);
  u += 0x7FFFu + ((u >> 16) & 1u);        // round-to-nearest-even
  return (unsigned short)(u >> 16);
}
static __device__ __forceinline__ float bf2f(unsigned short h) {
  unsigned int u = ((unsigned int)h) << 16;
  return __builtin_bit_cast(float, u);
}

// ---------------------------------------------------------------------------
// 1) Per-(h,n) discretization: w = dt*A, Ck = 2*(C_re+iC_im)*expm1(dtA)/A
// ---------------------------------------------------------------------------
__global__ void k_params(const float* __restrict__ log_dt,
                         const float* __restrict__ log_A_re,
                         const float* __restrict__ A_im,
                         const float* __restrict__ C_re,
                         const float* __restrict__ C_im,
                         float* __restrict__ wr_, float* __restrict__ wi_,
                         float* __restrict__ c0r_, float* __restrict__ c0i_,
                         float* __restrict__ c1r_, float* __restrict__ c1i_) {
  int i = blockIdx.x * blockDim.x + threadIdx.x;
  if (i >= Hdim * Ndim) return;
  int h = i / Ndim;
  float dt  = __expf(log_dt[h]);
  float Are = -__expf(log_A_re[i]);
  float Aim = A_im[i];
  float wr = Are * dt, wi = Aim * dt;
  float ew = __expf(wr);
  float s, c; __sincosf(wi, &s, &c);
  float Er = ew * c - 1.0f, Ei = ew * s;              // expm1(dtA)
  float inv = 1.0f / (Are * Are + Aim * Aim);
  float Fr = (Er * Are + Ei * Aim) * inv;             // expm1(dtA)/A
  float Fi = (Ei * Are - Er * Aim) * inv;
  float c0re = C_re[i],             c0im = C_im[i];
  float c1re = C_re[Hdim*Ndim + i], c1im = C_im[Hdim*Ndim + i];
  c0r_[i] = 2.0f * (c0re * Fr - c0im * Fi);
  c0i_[i] = 2.0f * (c0re * Fi + c0im * Fr);
  c1r_[i] = 2.0f * (c1re * Fr - c1im * Fi);
  c1i_[i] = 2.0f * (c1re * Fi + c1im * Fr);
  wr_[i] = wr; wi_[i] = wi;
}

// ---------------------------------------------------------------------------
// 2) Vandermonde kernel generation: K[c,h,l] = sum_n Re(Ck * exp(w*l))
// ---------------------------------------------------------------------------
__global__ void __launch_bounds__(256)
k_kern(const float* __restrict__ wr_, const float* __restrict__ wi_,
       const float* __restrict__ c0r_, const float* __restrict__ c0i_,
       const float* __restrict__ c1r_, const float* __restrict__ c1i_,
       float* __restrict__ K0, float* __restrict__ K1) {
  __shared__ float s[6 * Ndim];
  int h = blockIdx.x >> 3;
  int l = (blockIdx.x & 7) * 256 + threadIdx.x;
  int base = h * Ndim;
  if (threadIdx.x < Ndim) {
    int t = threadIdx.x;
    s[t]          = wr_[base + t];
    s[Ndim + t]   = wi_[base + t];
    s[2*Ndim + t] = c0r_[base + t];
    s[3*Ndim + t] = c0i_[base + t];
    s[4*Ndim + t] = c1r_[base + t];
    s[5*Ndim + t] = c1i_[base + t];
  }
  __syncthreads();
  float lf = (float)l;
  float k0 = 0.0f, k1 = 0.0f;
  #pragma unroll 4
  for (int n = 0; n < Ndim; ++n) {
    float a = __expf(s[n] * lf);
    float sn, cs; __sincosf(s[Ndim + n] * lf, &sn, &cs);
    k0 += a * (s[2*Ndim + n] * cs - s[3*Ndim + n] * sn);
    k1 += a * (s[4*Ndim + n] * cs - s[5*Ndim + n] * sn);
  }
  K0[(size_t)h * Ldim + l] = k0;
  K1[(size_t)h * Ldim + l] = k1;
}

// ---------------------------------------------------------------------------
// 3) W_out -> bf16
// ---------------------------------------------------------------------------
__global__ void k_w2bf(const float* __restrict__ W, unsigned short* __restrict__ Wbf) {
  int i = blockIdx.x * blockDim.x + threadIdx.x;
  if (i < Hdim * Hdim) Wbf[i] = f2bf_bits(W[i]);
}

// ---------------------------------------------------------------------------
// 4) Zero-fill batch pad rows 8..15 of zpad (16,H,L) bf16
// ---------------------------------------------------------------------------
__global__ void k_zpadfill(unsigned short* __restrict__ zpad) {
  // fill 8*H*L ushorts = 4M ushorts, 8 per thread via uint4
  int i = blockIdx.x * blockDim.x + threadIdx.x;   // 524288 threads
  uint4* p = (uint4*)(zpad + (size_t)Bdim * Hdim * Ldim);
  p[i] = make_uint4(0u, 0u, 0u, 0u);
}

// ---------------------------------------------------------------------------
// 5) LayerNorm over H (feature dim), store z as bf16 into rows 0..7 of zpad
// ---------------------------------------------------------------------------
__global__ void __launch_bounds__(256)
k_ln(const float* __restrict__ x, const float* __restrict__ ln_w,
     const float* __restrict__ ln_b, unsigned short* __restrict__ zpad) {
  int idx = blockIdx.x * blockDim.x + threadIdx.x;   // over B*L
  if (idx >= Bdim * Ldim) return;
  int b = idx / Ldim, l = idx % Ldim;
  const float* xp = x + (size_t)b * Hdim * Ldim + l;
  float sum = 0.0f, sq = 0.0f;
  for (int h = 0; h < Hdim; ++h) { float v = xp[(size_t)h * Ldim]; sum += v; sq += v * v; }
  float mu   = sum * (1.0f / Hdim);
  float var  = sq  * (1.0f / Hdim) - mu * mu;
  float rstd = rsqrtf(var + 1e-5f);
  for (int h = 0; h < Hdim; ++h) {
    float v = (xp[(size_t)h * Ldim] - mu) * rstd * ln_w[h] + ln_b[h];
    zpad[((size_t)b * Hdim + h) * Ldim + l] = f2bf_bits(v);
  }
}

// ---------------------------------------------------------------------------
// 6) Build Toeplitz A-tiles in WMMA fragment order.
//    Tile (h, didx) is the 16x32 bf16 A-matrix with delta = l0 - t0:
//      A[m][k] = K0[h, delta+m-k]        (delta+m-k >= 0)
//              = K1[h, -(delta+m-k)-1]   (otherwise)
//    stored as 512 bf16: offset = lane*16 + j, in this code's A fragment order
//      lane: m = lane&15, kh = lane>>4 ; k(j) = kh*16 + j (contiguous k order)
// ---------------------------------------------------------------------------
__global__ void __launch_bounds__(256)
k_atab(const float* __restrict__ K0, const float* __restrict__ K1,
       unsigned short* __restrict__ Atab) {
  int i = blockIdx.x * blockDim.x + threadIdx.x;     // Hdim*NDELTA*512
  if (i >= Hdim * NDELTA * 512) return;
  int e    = i & 511;
  int tile = i >> 9;
  int didx = tile % NDELTA;
  int h    = tile / NDELTA;
  int lane = e >> 4, j = e & 15;
  int m  = lane & 15, kh = lane >> 4;
  int kk = kh * 16 + j;
  int d  = (DELTA_MIN + didx * 16) + m - kk;         // in [-2047, 2047]
  const float* K0h = K0 + (size_t)h * Ldim;
  const float* K1h = K1 + (size_t)h * Ldim;
  float g = (d >= 0) ? K0h[d] : K1h[-d - 1];
  Atab[i] = f2bf_bits(g);
}

// ---------------------------------------------------------------------------
// 7) Toeplitz convolution via WMMA bf16. One wave per (h, 16-wide l-tile).
//    Inner loop: 2x b128 (A, fragment-order table) + 2x b128 (B, padded z)
//    + 1 WMMA. Epilogue: +z*D, exact GELU, store act transposed (B,L,H) bf16.
// ---------------------------------------------------------------------------
__global__ void __launch_bounds__(256)
k_conv(const unsigned short* __restrict__ zpad,
       const unsigned short* __restrict__ Atab,
       const float* __restrict__ Dp, unsigned short* __restrict__ act_t) {
  int lane = threadIdx.x & 31;
  int wave = blockIdx.x * (blockDim.x >> 5) + (threadIdx.x >> 5);
  int h  = wave >> 7;                  // 128 l-tiles per h
  int l0 = (wave & 127) * 16;
  int m  = lane & 15;                  // A row / B col / D col
  int kh = lane >> 4;
  const unsigned short* arow = Atab + (((size_t)h * NDELTA) << 9) + (lane << 4);
  const unsigned short* zrow = zpad + ((size_t)m * Hdim + h) * Ldim + kh * 16;

  v8f acc = {};
  for (int t0 = 0; t0 < Ldim; t0 += 32) {
    int didx = (l0 - t0 - DELTA_MIN) >> 4;
    v16bf a  = *(const v16bf*)(arow + ((size_t)didx << 9));
    v16bf bm = *(const v16bf*)(zrow + t0);
    acc = __builtin_amdgcn_wmma_f32_16x16x32_bf16(false, a, false, bm,
                                                  (short)0, acc, false, false);
  }

  float Dh = Dp[h];
  if (m < Bdim) {
    #pragma unroll
    for (int r = 0; r < 8; ++r) {
      int l  = l0 + r + kh * 8;
      float zf = bf2f(zpad[((size_t)m * Hdim + h) * Ldim + l]);
      float v  = acc[r] + zf * Dh;
      float g  = 0.5f * v * (1.0f + erff(v * 0.70710678118f));   // exact GELU
      act_t[((size_t)m * Ldim + l) * Hdim + h] = f2bf_bits(g);   // (B,L,H)
    }
  }
}

// ---------------------------------------------------------------------------
// 8) Output GEMM: out[b,o,l] = sum_h W[o,h]*act[b,h,l] + b_out[o] + x[b,o,l]
//    All fragment loads are aligned b128 vector loads.
// ---------------------------------------------------------------------------
__global__ void __launch_bounds__(256)
k_out(const unsigned short* __restrict__ Wbf,
      const unsigned short* __restrict__ act_t,
      const float* __restrict__ x, const float* __restrict__ b_out,
      float* __restrict__ out) {
  int lane = threadIdx.x & 31;
  int wave = blockIdx.x * (blockDim.x >> 5) + (threadIdx.x >> 5);
  int ot = wave & 15;                  // 16 o-tiles
  int nt = wave >> 4;                  // 1024 (b,l)-tiles
  int b  = nt >> 7;
  int l0 = (nt & 127) * 16;
  int m  = lane & 15;
  int kh = lane >> 4;
  int o  = ot * 16 + m;
  int l  = l0 + m;
  const unsigned short* wrow = Wbf   + (size_t)o * Hdim + kh * 16;     // 32B aligned
  const unsigned short* arow = act_t + ((size_t)b * Ldim + l) * Hdim + kh * 16; // 32B

  v8f acc = {};
  #pragma unroll
  for (int k0 = 0; k0 < Hdim; k0 += 32) {
    v8bf lo = *(const v8bf*)(wrow + k0);        // K = kh*16 + 0..7
    v8bf hi = *(const v8bf*)(wrow + k0 + 8);    // K = kh*16 + 8..15
    v16bf a;
    #pragma unroll
    for (int j = 0; j < 8; ++j) { a[j] = lo[j]; a[j + 8] = hi[j]; }
    v16bf bm = *(const v16bf*)(arow + k0);      // K = kh*16 + 0..15
    acc = __builtin_amdgcn_wmma_f32_16x16x32_bf16(false, a, false, bm,
                                                  (short)0, acc, false, false);
  }
  #pragma unroll
  for (int r = 0; r < 8; ++r) {
    int oo = ot * 16 + r + kh * 8;
    size_t oi = ((size_t)b * Hdim + oo) * Ldim + l;
    out[oi] = acc[r] + b_out[oo] + x[oi];
  }
}

// ---------------------------------------------------------------------------
extern "C" void kernel_launch(void* const* d_in, const int* in_sizes, int n_in,
                              void* d_out, int out_size, void* d_ws, size_t ws_size,
                              hipStream_t stream) {
  const float* x        = (const float*)d_in[0];
  const float* log_dt   = (const float*)d_in[1];
  const float* log_A_re = (const float*)d_in[2];
  const float* A_im     = (const float*)d_in[3];
  const float* C_re     = (const float*)d_in[4];
  const float* C_im     = (const float*)d_in[5];
  const float* Dp       = (const float*)d_in[6];
  const float* ln_w     = (const float*)d_in[7];
  const float* ln_b     = (const float*)d_in[8];
  const float* W_out    = (const float*)d_in[9];
  const float* b_out    = (const float*)d_in[10];
  float* out = (float*)d_out;

  char* ws = (char*)d_ws;
  size_t off = 0;
  auto carve = [&](size_t bytes) {
    void* p = ws + off;
    off = (off + bytes + 255) & ~(size_t)255;
    return p;
  };
  const size_t HN = (size_t)Hdim * Ndim, HL = (size_t)Hdim * Ldim;
  float* wr_  = (float*)carve(HN * 4);
  float* wi_  = (float*)carve(HN * 4);
  float* c0r_ = (float*)carve(HN * 4);
  float* c0i_ = (float*)carve(HN * 4);
  float* c1r_ = (float*)carve(HN * 4);
  float* c1i_ = (float*)carve(HN * 4);
  float* K0   = (float*)carve(HL * 4);
  float* K1   = (float*)carve(HL * 4);
  unsigned short* zpad  = (unsigned short*)carve((size_t)16 * HL * 2);      // 16 MB
  unsigned short* act_t = (unsigned short*)carve((size_t)Bdim * HL * 2);    // 8 MB
  unsigned short* Wbf   = (unsigned short*)carve((size_t)Hdim * Hdim * 2);
  unsigned short* Atab  = (unsigned short*)carve((size_t)Hdim * NDELTA * 512 * 2); // 65 MB

  k_params<<<(Hdim * Ndim) / 256, 256, 0, stream>>>(log_dt, log_A_re, A_im, C_re, C_im,
                                                    wr_, wi_, c0r_, c0i_, c1r_, c1i_);
  k_kern<<<Hdim * (Ldim / 256), 256, 0, stream>>>(wr_, wi_, c0r_, c0i_, c1r_, c1i_, K0, K1);
  k_w2bf<<<(Hdim * Hdim) / 256, 256, 0, stream>>>(W_out, Wbf);
  k_zpadfill<<<(Bdim * Hdim * Ldim / 8) / 256, 256, 0, stream>>>(zpad);
  k_ln<<<(Bdim * Ldim) / 256, 256, 0, stream>>>(x, ln_w, ln_b, zpad);
  k_atab<<<(Hdim * NDELTA * 512 + 255) / 256, 256, 0, stream>>>(K0, K1, Atab);
  // waves = H * (L/16) = 32768 ; 8 waves / block
  k_conv<<<(Hdim * (Ldim / 16)) / 8, 256, 0, stream>>>(zpad, Atab, Dp, act_t);
  // waves = 16 o-tiles * 1024 (b,l)-tiles = 16384 ; 8 waves / block
  k_out<<<(16 * Bdim * (Ldim / 16)) / 8, 256, 0, stream>>>(Wbf, act_t, x, b_out, out);
}
